// AttentionProtoNet_26336739459193
// MI455X (gfx1250) — matrix operand order repaired and checked by reading, not compiled, not run on hardware
//
#include <hip/hip_runtime.h>

// ---------- problem constants ----------
#define Bz 64
#define Lz 512
#define Hz 768
static_assert(Hz % 32 == 0, "");

typedef __attribute__((ext_vector_type(16))) __bf16    v16bf;
typedef __attribute__((ext_vector_type(8)))  float     v8f;
typedef __attribute__((ext_vector_type(8)))  unsigned  v8u;

// ---- packed bf16 helpers (avoid v_mov_b16 element shuffling) ----
__device__ __forceinline__ float hi_part(float x) {          // truncate to bf16-representable
  unsigned u = __builtin_bit_cast(unsigned, x) & 0xffff0000u;
  return __builtin_bit_cast(float, u);
}
__device__ __forceinline__ unsigned hi_pack(float x0, float x1) { // [x1.hi16 | x0.hi16]
  unsigned u0 = __builtin_bit_cast(unsigned, x0);
  unsigned u1 = __builtin_bit_cast(unsigned, x1);
  return (u0 >> 16) | (u1 & 0xffff0000u);
}
__device__ __forceinline__ unsigned bf16_bits_rne(float x) {  // RNE f32->bf16 bits
  unsigned u = __builtin_bit_cast(unsigned, x);
  u += 0x7fffu + ((u >> 16) & 1u);
  return u >> 16;
}
__device__ __forceinline__ unsigned lo_pack(float r0, float r1) {
  return (bf16_bits_rne(r0) & 0xffffu) | (bf16_bits_rne(r1) << 16);
}

// =====================================================================
// Kernel 1: nu[b,l] = sum_n W_nu[n] * tanh( hs[b,l,:] . W_fc[n,:] + b_fc[n] )
// bf16x3-split WMMA GEMM, f32 accumulate. 256 blocks x 256 threads.
// Wave owns 16 token rows; block stages 128-column W_fc chunks in LDS.
// =====================================================================
__global__ __launch_bounds__(256) void fused_fc_tanh_nu(
    const float* __restrict__ hs, const float* __restrict__ Wfc,
    const float* __restrict__ bfc, const float* __restrict__ Wnu,
    float* __restrict__ nu_out)
{
  constexpr int KT      = Hz / 32;     // 24 k-tiles
  constexpr int NCHUNK  = 128;         // columns staged per LDS chunk
  constexpr int NTILES  = NCHUNK / 16; // 8
  constexpr int NCHUNKS = Hz / NCHUNK; // 6

  // packed bf16 pairs: word index = n*16 + k/2   (k in [0,32))
  __shared__ __align__(32) unsigned sBhi[NCHUNK * 16];
  __shared__ __align__(32) unsigned sBlo[NCHUNK * 16];

  const int tid    = threadIdx.x;
  const int wave   = tid >> 5;
  const int lane   = tid & 31;
  const int lane15 = lane & 15;
  const int half   = lane >> 4;
  const int m0     = blockIdx.x * 128 + wave * 16;
  const float* arow = hs + (size_t)(m0 + lane15) * Hz;

  float nuacc[8] = {0.f,0.f,0.f,0.f,0.f,0.f,0.f,0.f};

  for (int nc = 0; nc < NCHUNKS; ++nc) {
    const int nbase = nc * NCHUNK;

    v8f c[NTILES];
    #pragma unroll
    for (int t = 0; t < NTILES; ++t)
      #pragma unroll
      for (int e = 0; e < 8; ++e) c[t][e] = 0.f;

    for (int kt = 0; kt < KT; ++kt) {
      const int k0 = kt * 32;

      __syncthreads();                   // previous stage fully consumed
      {
        // stage W_fc[nbase..+128, k0..+32]: thread -> one row, 16 k values
        const int nrel = tid >> 1;       // 0..127
        const int krel = (tid & 1) * 16; // 0 or 16
        const float* src = Wfc + (size_t)(nbase + nrel) * Hz + k0 + krel;
        if (kt + 1 < KT) __builtin_prefetch(src + 32, 0, 3);
        float4 w0 = *(const float4*)(src);
        float4 w1 = *(const float4*)(src + 4);
        float4 w2 = *(const float4*)(src + 8);
        float4 w3 = *(const float4*)(src + 12);
        float wv[16] = {w0.x,w0.y,w0.z,w0.w, w1.x,w1.y,w1.z,w1.w,
                        w2.x,w2.y,w2.z,w2.w, w3.x,w3.y,w3.z,w3.w};
        unsigned ph[8], pl[8];
        #pragma unroll
        for (int w = 0; w < 8; ++w) {
          float x0 = wv[2*w], x1 = wv[2*w+1];
          ph[w] = hi_pack(x0, x1);
          pl[w] = lo_pack(x0 - hi_part(x0), x1 - hi_part(x1));
        }
        const int wbase = nrel*16 + (krel >> 1);      // multiple of 8
        *(uint4*)&sBhi[wbase]     = make_uint4(ph[0],ph[1],ph[2],ph[3]);
        *(uint4*)&sBhi[wbase + 4] = make_uint4(ph[4],ph[5],ph[6],ph[7]);
        *(uint4*)&sBlo[wbase]     = make_uint4(pl[0],pl[1],pl[2],pl[3]);
        *(uint4*)&sBlo[wbase + 4] = make_uint4(pl[4],pl[5],pl[6],pl[7]);
      }
      __syncthreads();                   // stage visible to all waves

      // ---- A fragment (16x32 bf16 A-layout): row = lane&15,
      //      elems 0..15 -> K = k0+16*half+j
      const float* ap = arow + k0 + 16*half;
      float4 a0 = *(const float4*)(ap);
      float4 a1 = *(const float4*)(ap + 4);
      float4 a2 = *(const float4*)(ap + 8);
      float4 a3 = *(const float4*)(ap + 12);
      float av[16] = {a0.x,a0.y,a0.z,a0.w, a1.x,a1.y,a1.z,a1.w,
                      a2.x,a2.y,a2.z,a2.w, a3.x,a3.y,a3.z,a3.w};
      v8u ahw, alw;
      #pragma unroll
      for (int w = 0; w < 8; ++w) {
        float x0 = av[2*w], x1 = av[2*w+1];
        ahw[w] = hi_pack(x0, x1);
        alw[w] = lo_pack(x0 - hi_part(x0), x1 - hi_part(x1));
      }
      v16bf ahi = __builtin_bit_cast(v16bf, ahw);
      v16bf alo = __builtin_bit_cast(v16bf, alw);

      // ---- B fragments (32x16 B-layout: lanes 0-15 K=0..15, 16-31 K=16..31;
      //      lane&15 = N) read as 8 packed words, then 3 split WMMAs
      #pragma unroll
      for (int t = 0; t < NTILES; ++t) {
        const int woff = (t*16 + lane15)*16 + 8*half;   // 32B aligned
        v16bf bhi = __builtin_bit_cast(v16bf, *(const v8u*)&sBhi[woff]);
        v16bf blo = __builtin_bit_cast(v16bf, *(const v8u*)&sBlo[woff]);
        c[t] = __builtin_amdgcn_wmma_f32_16x16x32_bf16(
                   false, ahi, false, bhi, (short)0, c[t], false, false);
        c[t] = __builtin_amdgcn_wmma_f32_16x16x32_bf16(
                   false, alo, false, bhi, (short)0, c[t], false, false);
        c[t] = __builtin_amdgcn_wmma_f32_16x16x32_bf16(
                   false, ahi, false, blo, (short)0, c[t], false, false);
      }
    }

    // ---- epilogue: bias + tanh + dot(W_nu) for this 128-column chunk.
    // C layout: VGPR r, lanes 0-15 -> (M=r, N=lane15); lanes 16-31 -> (M=r+8)
    #pragma unroll
    for (int t = 0; t < NTILES; ++t) {
      const int n  = nbase + t*16 + lane15;
      const float bn = bfc[n];
      const float wn = Wnu[n];
      #pragma unroll
      for (int r = 0; r < 8; ++r)
        nuacc[r] += tanhf(c[t][r] + bn) * wn;
    }
  }

  // reduce nu across the 16-lane N groups (xor 1,2,4,8 stays within half)
  #pragma unroll
  for (int r = 0; r < 8; ++r) {
    float v = nuacc[r];
    v += __shfl_xor(v, 1, 32);
    v += __shfl_xor(v, 2, 32);
    v += __shfl_xor(v, 4, 32);
    v += __shfl_xor(v, 8, 32);
    nuacc[r] = v;
  }
  if (lane15 == 0) {
    #pragma unroll
    for (int r = 0; r < 8; ++r)
      nu_out[m0 + r + 8*half] = nuacc[r];
  }
}

// =====================================================================
// Kernel 2: softmax over L per batch, then pooled[b,h] = sum_l a_l*hs[b,l,h]
// =====================================================================
__global__ __launch_bounds__(256) void softmax_pool(
    const float* __restrict__ hs, const float* __restrict__ nu,
    float* __restrict__ out)
{
  __shared__ float s_alpha[Lz];
  __shared__ float s_rmax[8];
  __shared__ float s_rsum[8];

  const int b = blockIdx.x, tid = threadIdx.x;
  const float* nub = nu + b * Lz;

  float v0 = nub[tid], v1 = nub[tid + 256];
  float m = fmaxf(v0, v1);
  #pragma unroll
  for (int off = 16; off >= 1; off >>= 1) m = fmaxf(m, __shfl_xor(m, off, 32));
  if ((tid & 31) == 0) s_rmax[tid >> 5] = m;
  __syncthreads();

  float mall = s_rmax[0];
  #pragma unroll
  for (int w = 1; w < 8; ++w) mall = fmaxf(mall, s_rmax[w]);

  float e0 = __expf(v0 - mall), e1 = __expf(v1 - mall);
  s_alpha[tid] = e0; s_alpha[tid + 256] = e1;
  float s = e0 + e1;
  #pragma unroll
  for (int off = 16; off >= 1; off >>= 1) s += __shfl_xor(s, off, 32);
  if ((tid & 31) == 0) s_rsum[tid >> 5] = s;
  __syncthreads();

  float sall = 0.f;
  #pragma unroll
  for (int w = 0; w < 8; ++w) sall += s_rsum[w];
  const float inv = 1.0f / sall;

  const float* base = hs + (size_t)b * Lz * Hz;
  float acc0 = 0.f, acc1 = 0.f, acc2 = 0.f;
  #pragma unroll 4
  for (int l = 0; l < Lz; ++l) {
    const float a  = s_alpha[l];
    const float* r = base + (size_t)l * Hz + tid;
    acc0 = fmaf(a, r[0],   acc0);
    acc1 = fmaf(a, r[256], acc1);
    acc2 = fmaf(a, r[512], acc2);
  }
  float* ob = out + b * Hz + tid;
  ob[0]   = acc0 * inv;
  ob[256] = acc1 * inv;
  ob[512] = acc2 * inv;
}

// =====================================================================
extern "C" void kernel_launch(void* const* d_in, const int* in_sizes, int n_in,
                              void* d_out, int out_size, void* d_ws, size_t ws_size,
                              hipStream_t stream) {
  (void)in_sizes; (void)n_in; (void)out_size; (void)ws_size;
  const float* hs  = (const float*)d_in[0];   // [64,512,768]
  const float* Wfc = (const float*)d_in[1];   // [768,768]
  const float* bfc = (const float*)d_in[2];   // [768]
  const float* Wnu = (const float*)d_in[3];   // [768]
  float* out = (float*)d_out;                 // [64,768]
  float* nu  = (float*)d_ws;                  // [64*512] scratch logits

  fused_fc_tanh_nu<<<256, 256, 0, stream>>>(hs, Wfc, bfc, Wnu, nu);
  softmax_pool<<<Bz, 256, 0, stream>>>(hs, nu, out);
}
